// ForwardSim_12180527251945
// MI455X (gfx1250) — matrix operand order repaired and checked by reading, not compiled, not run on hardware
//
#include <hip/hip_runtime.h>
#include <hip/hip_bf16.h>

typedef __attribute__((ext_vector_type(16))) __bf16 v16bf;
typedef __attribute__((ext_vector_type(8)))  __bf16 v8bf;
typedef __attribute__((ext_vector_type(8)))  float  v8f;

namespace {
constexpr int kT           = 50;
constexpr int kProj        = 64;
constexpr int kInDim       = 73;   // 64 + 6 + 3
constexpr int kH           = 128;
constexpr int kNT          = 8;    // 128 / 16 N-tiles
constexpr int kK1          = 3;    // K padded 73 -> 96 -> 3 k-steps of 32
constexpr int kK2          = 4;    // 128 / 32
constexpr int kWaves       = 2;
constexpr int kThreads     = 32 * kWaves;
constexpr int kRowsPerWave = 16;
constexpr int kRowsPerBlk  = kWaves * kRowsPerWave;
}

__device__ __forceinline__ v8f wmma_bf16(v16bf a, v16bf b, v8f c) {
  // D = A(16x32 bf16) x B(32x16 bf16) + C(16x16 f32)
  return __builtin_amdgcn_wmma_f32_16x16x32_bf16(false, a, false, b,
                                                 (short)0, c, false, false);
}

// A-fragment from a row-major 16x128 bf16 LDS stage.
// Lane layout (16-bit A 16x32): halves 0..7 = K kb..kb+7, halves 8..15 = K kb+16..kb+23,
// with kb = ks*32 + (lane>=16 ? 8 : 0), row M = lane&15.
__device__ __forceinline__ v16bf load_a_frag(const __bf16* rowPtr, int kb) {
  v8bf c0 = *(const v8bf*)(rowPtr + kb);
  v8bf c1 = *(const v8bf*)(rowPtr + kb + 16);
  return __builtin_shufflevector(c0, c1, 0,1,2,3,4,5,6,7,8,9,10,11,12,13,14,15);
}

__global__ __launch_bounds__(kThreads)
void fused_rollout_mlp(const float* __restrict__ proj,
                       const float* __restrict__ idm,
                       const float* __restrict__ merg,
                       const float* __restrict__ W1,
                       const float* __restrict__ b1,
                       const float* __restrict__ W2,
                       const float* __restrict__ b2,
                       const float* __restrict__ W3,
                       const float* __restrict__ b3,
                       const float* __restrict__ smean,
                       const float* __restrict__ svar,
                       float* __restrict__ out)
{
  // B-fragment swizzled weights: [tile][lane][16 contiguous K-halves] -> 32B/lane loads
  __shared__ __align__(32) __bf16 sW1[kK1 * kNT][32][16];  // 24 KB
  __shared__ __align__(32) __bf16 sW2[kK2 * kNT][32][16];  // 32 KB
  __shared__ __align__(32) __bf16 sH [kWaves][16][kH];     //  8 KB (h1 stage + act bounce)
  // total static LDS = 65536 B exactly

  const int tid = threadIdx.x;

  // ---- one-time: swizzle W1 (73x128 f32, K zero-padded to 96) into B-fragment layout
  for (int i = tid; i < kK1 * kNT * 512; i += kThreads) {
    const int tile = i >> 9, rem = i & 511;
    const int ln = rem >> 4, hh = rem & 15;
    const int ks = tile / kNT, nt = tile % kNT;
    const int k = ks * 32 + ((ln & 16) ? 8 : 0) + (hh & 7) + ((hh & 8) ? 16 : 0);
    const int n = nt * 16 + (ln & 15);
    sW1[tile][ln][hh] = (k < kInDim) ? (__bf16)W1[k * kH + n] : (__bf16)0.0f;
  }
  // ---- one-time: swizzle W2 (128x128 f32)
  for (int i = tid; i < kK2 * kNT * 512; i += kThreads) {
    const int tile = i >> 9, rem = i & 511;
    const int ln = rem >> 4, hh = rem & 15;
    const int ks = tile / kNT, nt = tile % kNT;
    const int k = ks * 32 + ((ln & 16) ? 8 : 0) + (hh & 7) + ((hh & 8) ? 16 : 0);
    const int n = nt * 16 + (ln & 15);
    sW2[tile][ln][hh] = (__bf16)W2[k * kH + n];
  }
  __syncthreads();

  const int lane = tid & 31;
  const int wave = tid >> 5;
  const int col  = lane & 15;
  const bool hi  = (lane & 16) != 0;
  const int khi8 = hi ? 8 : 0;
  const int row  = blockIdx.x * kRowsPerBlk + wave * kRowsPerWave + col;

  // per-lane bias / W3 registers: C/D layout column N = nt*16 + col
  float b1r[kNT], b2r[kNT], w3r[kNT];
#pragma unroll
  for (int nt = 0; nt < kNT; ++nt) {
    b1r[nt] = b1[nt * 16 + col];
    b2r[nt] = b2[nt * 16 + col];
    w3r[nt] = W3[nt * 16 + col];
  }
  const float b3s = b3[0];

  float mean_[6], istd_[6];
#pragma unroll
  for (int i = 0; i < 6; ++i) { mean_[i] = smean[i]; istd_[i] = 1.0f / sqrtf(svar[i]); }

  // loop-invariant A fragments for proj_belief (K = 0..63, k-steps 0 and 1)
  v16bf aProj[2];
#pragma unroll
  for (int ks = 0; ks < 2; ++ks) {
    const float* p = proj + (size_t)row * kProj + ks * 32 + khi8;
#pragma unroll
    for (int h = 0; h < 8; ++h) {
      aProj[ks][h]     = (__bf16)p[h];
      aProj[ks][8 + h] = (__bf16)p[16 + h];
    }
  }

  const __bf16* hrow = &sH[wave][col][0];
  float* actS = (float*)(&sH[wave][0][0]);   // 64B act bounce aliased into h1 stage

  // carried pointers (avoid re-deriving row*kT*stride addressing every step)
  const float* sP  = idm  + (size_t)row * kT * 12;
  const float* mcP = merg + (size_t)row * kT * 3;
  float*       oP  = out  + (size_t)row * kT;

  float ego_v = 0.0f, ego_x = 0.0f, act = 0.0f;

#pragma unroll 1
  for (int t = 0; t < kT; ++t, sP += 12, mcP += 3, ++oP) {
    const float s0 = sP[0], s1 = sP[1], s2 = sP[2], s3 = sP[3], s4 = sP[4], s5 = sP[5],
                s11 = sP[11];
    const float m0 = mcP[0], m1 = mcP[1], m2 = mcP[2];

    // prefetch next step's per-row inputs into near caches (WGP scope).
    // clamp at rollout end: WGP-scope prefetch is non-speculative, keep it in-bounds.
    {
      const float* pfS  = (t + 1 < kT) ? sP + 12 : sP;
      const float* pfMc = (t + 1 < kT) ? mcP + 3 : mcP;
      __builtin_prefetch(pfS, 0, 3);
      __builtin_prefetch(pfMc, 0, 3);
    }

    // carry update (uniform branch on t)
    const float nv = ego_v + act * 0.1f;
    const float nx = ego_x + nv * 0.1f + 0.5f * act * 0.01f;
    ego_v = (t == 0) ? s0 : nv;
    ego_x = (t == 0) ? s3 : nx;

    const float e0 = (ego_v                                           - mean_[0]) * istd_[0];
    const float e1 = (s1                                              - mean_[1]) * istd_[1];
    const float e2 = ((ego_v - s1)                                    - mean_[2]) * istd_[2];
    const float e3 = ((s4 - ego_x)                                    - mean_[3]) * istd_[3];
    const float e4 = ((ego_v - s2) * s11                              - mean_[4]) * istd_[4];
    const float e5 = ((s5 - ego_x) * s11 + (1.0f - s11) * 100.0f      - mean_[5]) * istd_[5];

    // dynamic A fragment, k-step 2 (K=64..95): lanes<16 see K 64..71 = env0..5,mc0,mc1;
    // lanes>=16 see K 72..79 = mc2,0,...; halves 8..15 (K 80..95) are zero pad.
    v16bf aDyn;
    aDyn[0] = (__bf16)(hi ? m2 : e0);
    aDyn[1] = (__bf16)(hi ? 0.0f : e1);
    aDyn[2] = (__bf16)(hi ? 0.0f : e2);
    aDyn[3] = (__bf16)(hi ? 0.0f : e3);
    aDyn[4] = (__bf16)(hi ? 0.0f : e4);
    aDyn[5] = (__bf16)(hi ? 0.0f : e5);
    aDyn[6] = (__bf16)(hi ? 0.0f : m0);
    aDyn[7] = (__bf16)(hi ? 0.0f : m1);
#pragma unroll
    for (int h = 8; h < 16; ++h) aDyn[h] = (__bf16)0.0f;

    const v8f zeroC = 0.0f;   // splat-0 C -> backend encodes WMMA SRC2 as inline 0

    // ---- layer 1: h1 = relu(inp @ W1 + b1); C starts at inline 0, bias in epilogue
#pragma unroll
    for (int nt = 0; nt < kNT; ++nt) {
      v8f c;
      c = wmma_bf16(aProj[0], *(const v16bf*)(&sW1[0 * kNT + nt][lane][0]), zeroC);
      c = wmma_bf16(aProj[1], *(const v16bf*)(&sW1[1 * kNT + nt][lane][0]), c);
      c = wmma_bf16(aDyn,     *(const v16bf*)(&sW1[2 * kNT + nt][lane][0]), c);
#pragma unroll
      for (int r = 0; r < 8; ++r)
        sH[wave][r + khi8][nt * 16 + col] = (__bf16)fmaxf(c[r] + b1r[nt], 0.0f);
    }

    // ---- layer 2 + fused layer 3 epilogue:
    //      h2 = relu(h1 @ W2 + b2); part[r] += h2[M=r+khi8][N] * W3[N] (lane-local N)
    v16bf a2[kK2];
#pragma unroll
    for (int ks = 0; ks < kK2; ++ks) a2[ks] = load_a_frag(hrow, ks * 32 + khi8);

    float part[8];
#pragma unroll
    for (int r = 0; r < 8; ++r) part[r] = 0.0f;

#pragma unroll
    for (int nt = 0; nt < kNT; ++nt) {
      v8f c;
      c = wmma_bf16(a2[0], *(const v16bf*)(&sW2[0 * kNT + nt][lane][0]), zeroC);
#pragma unroll
      for (int ks = 1; ks < kK2; ++ks)
        c = wmma_bf16(a2[ks], *(const v16bf*)(&sW2[ks * kNT + nt][lane][0]), c);
#pragma unroll
      for (int r = 0; r < 8; ++r)
        part[r] = fmaf(fmaxf(c[r] + b2r[nt], 0.0f), w3r[nt], part[r]);
    }

    // butterfly-reduce part[] across each 16-lane half (masks 1,2,4,8 stay within halves):
    // low half -> act rows 0..7, high half -> act rows 8..15
#pragma unroll
    for (int off = 1; off < 16; off <<= 1) {
#pragma unroll
      for (int r = 0; r < 8; ++r)
        part[r] += __shfl_xor(part[r], off, 32);
    }

    // bounce act to owning lanes through LDS (rows 0..7 from lane 0, 8..15 from lane 16)
    if (lane == 0) {
#pragma unroll
      for (int r = 0; r < 8; ++r) actS[r] = part[r] + b3s;
    } else if (lane == 16) {
#pragma unroll
      for (int r = 0; r < 8; ++r) actS[8 + r] = part[r] + b3s;
    }
    act = actS[col];   // same-wave DS ops are in-order; compiler inserts s_wait_dscnt

    if (!hi) *oP = act;
  }
}

extern "C" void kernel_launch(void* const* d_in, const int* in_sizes, int n_in,
                              void* d_out, int out_size, void* d_ws, size_t ws_size,
                              hipStream_t stream) {
  const float* proj = (const float*)d_in[0];
  const float* idm  = (const float*)d_in[1];
  const float* merg = (const float*)d_in[2];
  const float* W1   = (const float*)d_in[3];
  const float* b1   = (const float*)d_in[4];
  const float* W2   = (const float*)d_in[5];
  const float* b2   = (const float*)d_in[6];
  const float* W3   = (const float*)d_in[7];
  const float* b3   = (const float*)d_in[8];
  const float* sm   = (const float*)d_in[9];
  const float* sv   = (const float*)d_in[10];
  float* out = (float*)d_out;

  const int batch = in_sizes[0] / kProj;         // 32768
  dim3 grid(batch / kRowsPerBlk);                // 1024 blocks of 64 threads (2 waves)
  fused_rollout_mlp<<<grid, kThreads, 0, stream>>>(proj, idm, merg, W1, b1, W2, b2,
                                                   W3, b3, sm, sv, out);
}
